// LTC_sequence_50543175140034
// MI455X (gfx1250) — compile-verified, matrix operand order checked by reading, not run on hardware
//
#include <hip/hip_runtime.h>

// ---------------------------------------------------------------------------
// LTC sequence model for MI455X (gfx1250, wave32, WMMA)
//   encoder/decoder MLPs : bf16 WMMA GEMMs, 8 WMMA per wave per k-chunk
//   LTC scan             : latency-bound; LDS-resident params, 4-way split
//   staging              : GLOBAL_LOAD_ASYNC_TO_LDS_B128 (ASYNCcnt) when avail
// ---------------------------------------------------------------------------

#define B_DIM     32
#define T_STEPS   512
#define M_ROWS    (B_DIM * T_STEPS)   // 16384
#define UNITS     64
#define SENS      20
#define UNFOLDS   6

typedef __attribute__((ext_vector_type(16))) __bf16 v16bf;
typedef __attribute__((ext_vector_type(8)))  float  v8f;
struct U16x16 { uint4 lo, hi; };

#if defined(__HIP_DEVICE_COMPILE__) && \
    __has_builtin(__builtin_amdgcn_global_load_async_to_lds_b128) && \
    __has_builtin(__builtin_amdgcn_s_wait_asynccnt)
#define USE_ASYNC_LDS 1
#else
#define USE_ASYNC_LDS 0
#endif

#if USE_ASYNC_LDS
typedef __attribute__((__vector_size__(4 * sizeof(int)))) int vs4i;
typedef __attribute__((address_space(1))) vs4i* as1_v4i_ptr;   // global
typedef __attribute__((address_space(3))) vs4i* as3_v4i_ptr;   // LDS
#endif

static __device__ __forceinline__ void cp16_to_lds(unsigned short* lds,
                                                   const unsigned short* g) {
#if USE_ASYNC_LDS
    __builtin_amdgcn_global_load_async_to_lds_b128(
        (as1_v4i_ptr)g, (as3_v4i_ptr)lds, 0, 0);
#else
    *(uint4*)lds = *(const uint4*)g;
#endif
}
static __device__ __forceinline__ void cp16_wait() {
#if USE_ASYNC_LDS
    __builtin_amdgcn_s_wait_asynccnt(0);
#endif
}

static __device__ __forceinline__ unsigned short f2bf(float f) {
    unsigned int u = __float_as_uint(f);
    u += 0x7FFFu + ((u >> 16) & 1u);          // round-to-nearest-even
    return (unsigned short)(u >> 16);
}
static __device__ __forceinline__ float fast_rcp(float x) { return __builtin_amdgcn_rcpf(x); }
static __device__ __forceinline__ float fast_exp2(float x){ return __builtin_amdgcn_exp2f(x); }
static __device__ __forceinline__ float fast_sigmoid(float x) {
    return fast_rcp(1.0f + fast_exp2(-1.4426950408889634f * x));
}
static __device__ __forceinline__ float fast_tanh(float x) {
    x = fminf(10.0f, fmaxf(-10.0f, x));
    float e = fast_exp2(2.8853900817779268f * x);   // exp(2x)
    return (e - 1.0f) * fast_rcp(e + 1.0f);
}

// --------------------------- conversion kernels ----------------------------
// x: [R,C] f32 -> [R,Cp] bf16 zero-padded (row-major, K-contiguous)
__global__ void pad_f32_to_bf16(const float* __restrict__ src,
                                unsigned short* __restrict__ dst,
                                int R, int C, int Cp, int total) {
    int i = blockIdx.x * blockDim.x + threadIdx.x;
    if (i >= total) return;
    int r = i / Cp, c = i % Cp;
    float v = (r < R && c < C) ? src[r * C + c] : 0.0f;
    dst[i] = f2bf(v);
}

// W: [K,N] f32 -> Wt: [Np,Kp] bf16 zero-padded (N-major, K-contiguous rows)
__global__ void pad_transpose_f32_to_bf16(const float* __restrict__ src,
                                          unsigned short* __restrict__ dst,
                                          int K, int N, int Kp, int total) {
    int i = blockIdx.x * blockDim.x + threadIdx.x;
    if (i >= total) return;
    int n = i / Kp, k = i % Kp;
    float v = (k < K && n < N) ? src[k * N + n] : 0.0f;
    dst[i] = f2bf(v);
}

__global__ void zero_u16(unsigned short* __restrict__ p, int n) {
    int i = blockIdx.x * blockDim.x + threadIdx.x;
    if (i < n) p[i] = 0;
}

// Fragment gather helpers (ISA 7.12.2 16-bit 16x32 layout):
// lanes 0-15: K 0..7 / 16..23 ; lanes 16-31: K 8..15 / 24..31
static __device__ __forceinline__ v16bf frag16(const unsigned short* row, int base) {
    U16x16 t;
    t.lo = *(const uint4*)(row + base);
    t.hi = *(const uint4*)(row + 16 + base);
    return __builtin_bit_cast(v16bf, t);
}

// ------------------------ bf16 WMMA GEMM (wide N) --------------------------
// C[M, Np] = tanh(A[M, Kp] @ W + bias)  with W pre-transposed as Wt[Np, Kp].
// 256 threads = 8 waves as 2(M) x 4(N); block tile 64(M) x 256(N);
// wave tile 32(M) x 64(N) = 8 WMMAs per k-chunk.
__global__ __launch_bounds__(256)
void gemm_bf16_wmma_wide(const unsigned short* __restrict__ A, int lda,
                         const unsigned short* __restrict__ Wt, int Kp, int Np,
                         const float* __restrict__ bias, int Norig,
                         unsigned short* __restrict__ outb) {
    __shared__ __align__(16) unsigned short lA [64  * 32];  // [m][k]
    __shared__ __align__(16) unsigned short lBt[256 * 32];  // [n][k]

    const int tid   = threadIdx.x;
    const int lane  = tid & 31;
    const int wid   = tid >> 5;
    const int waveM = wid >> 2;          // 0..1
    const int waveN = wid & 3;           // 0..3
    const int mBlock = blockIdx.x * 64;
    const int nBlock = blockIdx.y * 256;

    v8f acc[2][4];
#pragma unroll
    for (int mi = 0; mi < 2; ++mi)
#pragma unroll
        for (int ni = 0; ni < 4; ++ni)
            acc[mi][ni] = (v8f){0.f,0.f,0.f,0.f,0.f,0.f,0.f,0.f};

    const int aRow = tid >> 2, aPart = tid & 3;   // 64 rows x 4x(8 bf16)
    const int base = (lane & 16) ? 8 : 0;
    const int lrow = lane & 15;

    for (int kc = 0; kc < Kp; kc += 32) {
        { // A tile 64x32
            const unsigned short* gp = A + (mBlock + aRow) * lda + kc + aPart * 8;
            cp16_to_lds(&lA[aRow * 32 + aPart * 8], gp);
            if (kc + 32 < Kp) __builtin_prefetch(gp + 32, 0, 3);
        }
        { // B tile 256 rows(N) x 32(K), K-contiguous in global
            const unsigned short* gp = Wt + (nBlock + tid) * Kp + kc;
#pragma unroll
            for (int p = 0; p < 4; ++p)
                cp16_to_lds(&lBt[tid * 32 + p * 8], gp + p * 8);
            if (kc + 32 < Kp) __builtin_prefetch(gp + 32, 0, 3);
        }
        cp16_wait();
        __syncthreads();

        v16bf af[2], bf[4];
#pragma unroll
        for (int mi = 0; mi < 2; ++mi)
            af[mi] = frag16(&lA[(waveM * 32 + mi * 16 + lrow) * 32], base);
#pragma unroll
        for (int ni = 0; ni < 4; ++ni)
            bf[ni] = frag16(&lBt[(waveN * 64 + ni * 16 + lrow) * 32], base);
#pragma unroll
        for (int mi = 0; mi < 2; ++mi)
#pragma unroll
            for (int ni = 0; ni < 4; ++ni)
                acc[mi][ni] = __builtin_amdgcn_wmma_f32_16x16x32_bf16(
                    false, af[mi], false, bf[ni], (short)0, acc[mi][ni],
                    false, false);
        __syncthreads();
    }

    // C layout: VGPR r -> M = r + (lane>=16 ? 8 : 0), N = lane%16
#pragma unroll
    for (int ni = 0; ni < 4; ++ni) {
        const int n  = nBlock + waveN * 64 + ni * 16 + lrow;
        const float bv = (n < Norig) ? bias[n] : 0.0f;
#pragma unroll
        for (int mi = 0; mi < 2; ++mi) {
            const int mBase = mBlock + waveM * 32 + mi * 16 + ((lane & 16) ? 8 : 0);
#pragma unroll
            for (int r = 0; r < 8; ++r)
                outb[(mBase + r) * Np + n] = f2bf(fast_tanh(acc[mi][ni][r] + bv));
        }
    }
}

// ------------------------ bf16 WMMA GEMM (thin N) --------------------------
// Final layers: Np == 32, no activation, f32 output (n < Norig), ld = ldo.
// Block tile 64(M) x 32(N); 8 waves as 4(M) x 2(N), one WMMA tile each.
__global__ __launch_bounds__(256)
void gemm_bf16_wmma_thin(const unsigned short* __restrict__ A, int lda,
                         const unsigned short* __restrict__ Wt, int Kp,
                         const float* __restrict__ bias, int Norig,
                         float* __restrict__ outf, int ldo) {
    __shared__ __align__(16) unsigned short lA [64 * 32];
    __shared__ __align__(16) unsigned short lBt[32 * 32];

    const int tid   = threadIdx.x;
    const int lane  = tid & 31;
    const int wid   = tid >> 5;
    const int warpM = wid >> 1;          // 0..3
    const int warpN = wid & 1;           // 0..1
    const int mBlock = blockIdx.x * 64;

    v8f c = {0.f,0.f,0.f,0.f,0.f,0.f,0.f,0.f};
    const int aRow = tid >> 2, aPart = tid & 3;
    const int base = (lane & 16) ? 8 : 0;
    const int lrow = lane & 15;

    for (int kc = 0; kc < Kp; kc += 32) {
        {
            const unsigned short* gp = A + (mBlock + aRow) * lda + kc + aPart * 8;
            cp16_to_lds(&lA[aRow * 32 + aPart * 8], gp);
            if (kc + 32 < Kp) __builtin_prefetch(gp + 32, 0, 3);
        }
        if (tid < 128) { // 32 rows(N) x 4x(8 bf16)
            const int nr = tid >> 2, p = tid & 3;
            const unsigned short* gp = Wt + nr * Kp + kc + p * 8;
            cp16_to_lds(&lBt[nr * 32 + p * 8], gp);
            if (kc + 32 < Kp) __builtin_prefetch(gp + 32, 0, 3);
        }
        cp16_wait();
        __syncthreads();

        v16bf a = frag16(&lA [(warpM * 16 + lrow) * 32], base);
        v16bf b = frag16(&lBt[(warpN * 16 + lrow) * 32], base);
        c = __builtin_amdgcn_wmma_f32_16x16x32_bf16(false, a, false, b,
                                                    (short)0, c, false, false);
        __syncthreads();
    }

    const int mBase = mBlock + warpM * 16 + ((lane & 16) ? 8 : 0);
    const int n     = warpN * 16 + lrow;
    if (n < Norig) {
        const float bv = bias[n];
#pragma unroll
        for (int r = 0; r < 8; ++r)
            outf[(mBase + r) * ldo + n] = c[r] + bv;
    }
}

// ------------------------------- LTC scan ----------------------------------
// One block per batch element. 256 threads = (group g 0..3) x (unit j 0..63).
// Cell params resident in LDS; i-reduction split 4-way across groups.
__global__ __launch_bounds__(256)
void ltc_scan(const float* __restrict__ h,      // [B*T, 32] f32, first 20 valid
              const float* __restrict__ gleak, const float* __restrict__ vleak,
              const float* __restrict__ cm,
              const float* __restrict__ w,   const float* __restrict__ mu,
              const float* __restrict__ sg,  const float* __restrict__ er,
              const float* __restrict__ sw,  const float* __restrict__ smu,
              const float* __restrict__ ssg, const float* __restrict__ ser,
              const float* __restrict__ inw, const float* __restrict__ inb,
              const float* __restrict__ outw,const float* __restrict__ outb,
              unsigned short* __restrict__ motor) { // [B*T, 32] bf16, cols 0..3
    __shared__ float w_l[UNITS*UNITS], mu_l[UNITS*UNITS],
                     sg_l[UNITS*UNITS], er_l[UNITS*UNITS];
    __shared__ float sw_l[SENS*UNITS], smu_l[SENS*UNITS],
                     ssg_l[SENS*UNITS], ser_l[SENS*UNITS];
    __shared__ float v_l[UNITS], ui_l[SENS];
    __shared__ float pnum[4][UNITS], pden[4][UNITS];
    __shared__ float wns_l[UNITS], wds_l[UNITS];

    const int tid = threadIdx.x;
    const int j = tid & 63;      // unit
    const int g = tid >> 6;      // reduction group 0..3
    const int b = blockIdx.x;

    for (int i = tid; i < UNITS*UNITS; i += 256) {
        w_l[i] = w[i]; mu_l[i] = mu[i]; sg_l[i] = sg[i]; er_l[i] = er[i];
    }
    for (int i = tid; i < SENS*UNITS; i += 256) {
        sw_l[i] = sw[i]; smu_l[i] = smu[i]; ssg_l[i] = ssg[i]; ser_l[i] = ser[i];
    }
    const float gl  = gleak[j], vl = vleak[j];
    const float cmt = cm[j] * (float)UNFOLDS;      // cm / (1/6)
    float inwv = 0.f, inbv = 0.f, oww = 0.f, obb = 0.f;
    if (tid < SENS) { inwv = inw[tid]; inbv = inb[tid]; }
    if (tid < 4)    { oww = outw[tid]; obb = outb[tid]; }
    float vj = 0.0f;
    if (g == 0) v_l[j] = 0.0f;
    __syncthreads();

    const int rowBase = b * T_STEPS;
    for (int t = 0; t < T_STEPS; ++t) {
        if (tid < SENS) ui_l[tid] = h[(rowBase + t) * 32 + tid] * inwv + inbv;
        __syncthreads();

        { // sensory synapse partials: 5 sensory inputs per group
            float pn = 0.f, pd = 0.f;
            const int s0 = g * 5;
#pragma unroll
            for (int s = s0; s < s0 + 5; ++s) {
                int idx = s * UNITS + j;
                float sa = sw_l[idx] * fast_sigmoid((ui_l[s] - smu_l[idx]) * ssg_l[idx]);
                pn += sa * ser_l[idx];
                pd += sa;
            }
            pnum[g][j] = pn; pden[g][j] = pd;
        }
        __syncthreads();
        if (g == 0) {
            wns_l[j] = pnum[0][j] + pnum[1][j] + pnum[2][j] + pnum[3][j];
            wds_l[j] = pden[0][j] + pden[1][j] + pden[2][j] + pden[3][j];
        }
        __syncthreads();   // protect pnum/pden before unfold reuse

#pragma unroll 1
        for (int u = 0; u < UNFOLDS; ++u) {
            float pn = 0.f, pd = 0.f;
            const int i0 = g * 16;
#pragma unroll 4
            for (int i = i0; i < i0 + 16; ++i) {
                int idx = i * UNITS + j;
                float wa = w_l[idx] * fast_sigmoid((v_l[i] - mu_l[idx]) * sg_l[idx]);
                pn += wa * er_l[idx];
                pd += wa;
            }
            pnum[g][j] = pn; pden[g][j] = pd;
            __syncthreads();
            if (g == 0) {
                float num = wns_l[j] + pnum[0][j] + pnum[1][j] + pnum[2][j] + pnum[3][j];
                float den = wds_l[j] + pden[0][j] + pden[1][j] + pden[2][j] + pden[3][j];
                float vn = (cmt * vj + gl * vl + num) *
                           fast_rcp(cmt + gl + den + 1e-8f);
                v_l[j] = vn; vj = vn;
            }
            __syncthreads();
        }
        if (tid < 4) motor[(rowBase + t) * 32 + tid] = f2bf(vj * oww + obb);
    }
}

// ------------------------------- launch ------------------------------------
extern "C" void kernel_launch(void* const* d_in, const int* in_sizes, int n_in,
                              void* d_out, int out_size, void* d_ws, size_t ws_size,
                              hipStream_t stream) {
    const float* x      = (const float*)d_in[0];
    const float* enc_w0 = (const float*)d_in[1];
    const float* enc_b0 = (const float*)d_in[2];
    const float* enc_w1 = (const float*)d_in[3];
    const float* enc_b1 = (const float*)d_in[4];
    const float* enc_w2 = (const float*)d_in[5];
    const float* enc_b2 = (const float*)d_in[6];
    const float* dec_w0 = (const float*)d_in[7];
    const float* dec_b0 = (const float*)d_in[8];
    const float* dec_w1 = (const float*)d_in[9];
    const float* dec_b1 = (const float*)d_in[10];
    const float* dec_w2 = (const float*)d_in[11];
    const float* dec_b2 = (const float*)d_in[12];
    const float* gleak  = (const float*)d_in[13];
    const float* vleak  = (const float*)d_in[14];
    const float* cm     = (const float*)d_in[15];
    const float* wP     = (const float*)d_in[16];
    const float* muP    = (const float*)d_in[17];
    const float* sgP    = (const float*)d_in[18];
    const float* erP    = (const float*)d_in[19];
    const float* swP    = (const float*)d_in[20];
    const float* smuP   = (const float*)d_in[21];
    const float* ssgP   = (const float*)d_in[22];
    const float* serP   = (const float*)d_in[23];
    const float* in_w   = (const float*)d_in[24];
    const float* in_b   = (const float*)d_in[25];
    const float* out_w  = (const float*)d_in[26];
    const float* out_b  = (const float*)d_in[27];
    float* out = (float*)d_out;

    char* ws = (char*)d_ws;
    size_t off = 0;
    auto take = [&](size_t bytes) -> void* {
        void* p = ws + off;
        off = (off + bytes + 255) & ~(size_t)255;
        return p;
    };

    unsigned short* xbf   = (unsigned short*)take((size_t)M_ROWS * 64 * 2);
    unsigned short* w0t   = (unsigned short*)take((size_t)512 * 64  * 2);  // [512][64]
    unsigned short* w1t   = (unsigned short*)take((size_t)512 * 512 * 2);  // [512][512]
    unsigned short* w2t   = (unsigned short*)take((size_t)32  * 512 * 2);  // [32][512]
    unsigned short* dw0t  = (unsigned short*)take((size_t)512 * 32  * 2);  // [512][32]
    unsigned short* dw1t  = (unsigned short*)take((size_t)512 * 512 * 2);  // [512][512]
    unsigned short* dw2t  = (unsigned short*)take((size_t)32  * 512 * 2);  // [32][512]
    unsigned short* bufA  = (unsigned short*)take((size_t)M_ROWS * 512 * 2);
    unsigned short* bufB  = (unsigned short*)take((size_t)M_ROWS * 512 * 2);
    float*          henc  = (float*)         take((size_t)M_ROWS * 32 * 4);
    unsigned short* motor = (unsigned short*)take((size_t)M_ROWS * 32 * 2);

    // x -> padded bf16 (row-major, K-contiguous)
    {
        int total = M_ROWS * 64;
        pad_f32_to_bf16<<<dim3((total + 255) / 256), dim3(256), 0, stream>>>(
            x, xbf, M_ROWS, 39, 64, total);
    }
    // weights -> transposed+padded bf16: Wt[Np][Kp]
    auto padT = [&](const float* src, unsigned short* dst, int K, int N,
                    int Kp, int Np) {
        int total = Np * Kp;
        pad_transpose_f32_to_bf16<<<dim3((total + 255) / 256), dim3(256), 0,
                                    stream>>>(src, dst, K, N, Kp, total);
    };
    padT(enc_w0, w0t,  39,  500, 64,  512);
    padT(enc_w1, w1t,  500, 500, 512, 512);
    padT(enc_w2, w2t,  500, 20,  512, 32);
    padT(dec_w0, dw0t, 4,   500, 32,  512);
    padT(dec_w1, dw1t, 500, 500, 512, 512);
    padT(dec_w2, dw2t, 500, 30,  512, 32);

    {
        int n = M_ROWS * 32;
        zero_u16<<<dim3((n + 255) / 256), dim3(256), 0, stream>>>(motor, n);
    }

    auto gemmW = [&](const unsigned short* A, int lda, const unsigned short* Wt,
                     int Kp, const float* bias, int Norig, unsigned short* ob) {
        gemm_bf16_wmma_wide<<<dim3(M_ROWS / 64, 512 / 256), dim3(256), 0,
                              stream>>>(A, lda, Wt, Kp, 512, bias, Norig, ob);
    };
    auto gemmT = [&](const unsigned short* A, int lda, const unsigned short* Wt,
                     int Kp, const float* bias, int Norig, float* of, int ldo) {
        gemm_bf16_wmma_thin<<<dim3(M_ROWS / 64), dim3(256), 0, stream>>>(
            A, lda, Wt, Kp, bias, Norig, of, ldo);
    };

    // encoder
    gemmW(xbf,  64,  w0t, 64,  enc_b0, 500, bufA);
    gemmW(bufA, 512, w1t, 512, enc_b1, 500, bufB);
    gemmT(bufB, 512, w2t, 512, enc_b2, 20, henc, 32);

    // LTC scan: one block per batch element
    ltc_scan<<<dim3(B_DIM), dim3(256), 0, stream>>>(
        henc, gleak, vleak, cm, wP, muP, sgP, erP,
        swP, smuP, ssgP, serP, in_w, in_b, out_w, out_b, motor);

    // decoder
    gemmW(motor, 32, dw0t, 32,  dec_b0, 500, bufA);
    gemmW(bufA,  512, dw1t, 512, dec_b1, 500, bufB);
    gemmT(bufB,  512, dw2t, 512, dec_b2, 30, out, 30);

    (void)in_sizes; (void)n_in; (void)out_size; (void)ws_size;
}
